// MinimalHPM_30889404793177
// MI455X (gfx1250) — matrix operand order repaired and checked by reading, not compiled, not run on hardware
//
#include <hip/hip_runtime.h>

typedef __attribute__((ext_vector_type(16))) _Float16     v16h;
typedef __attribute__((ext_vector_type(8)))  float        v8f;
typedef __attribute__((ext_vector_type(4)))  unsigned int v4u;
typedef __attribute__((ext_vector_type(8)))  int          v8i;
typedef __attribute__((ext_vector_type(4)))  int          v4i;

#define DXX 128
#define DYY 128
#define DZZ 128
#define CCH 16
#define NVOX (DXX*DYY*DZZ)      // 2,097,152 voxels
#define BRAYS 64
#define KCHUNK 128              // one full z-line per chunk -> (x,y) constant
#define NCHUNKS (NVOX / KCHUNK) // 16384

// ---------------------------------------------------------------------------
// TDM: issue one tensor_load_to_lds moving a [128 voxels x 16 ch] f32 tile
// (8 KB) from global memory into LDS at lds_off. 2-D descriptor:
//   dim0 = 16 elements (channels, contiguous), dim1 = 128 rows, stride0 = 16.
// Tracked by TENSORcnt; EXEC is ignored, so issue from one wave only.
// ---------------------------------------------------------------------------
__device__ __forceinline__ void tdm_load_tile(const float* gsrc, unsigned lds_off)
{
    const unsigned long long ga = (unsigned long long)(uintptr_t)gsrc;

    v4u g0;
    g0[0] = 1u;                                   // count=1, user descriptor
    g0[1] = lds_off;                              // lds_addr (bytes)
    g0[2] = (unsigned)(ga & 0xFFFFFFFFu);         // global_addr[31:0]
    g0[3] = (unsigned)((ga >> 32) & 0x1FFFFFFu)   // global_addr[56:32]
          | (2u << 30);                           // type = 2 ("image")

    v8i g1;
    g1[0] = (2 << 16);          // data_size = 2 (4 bytes); workgroup_mask = 0
    g1[1] = (16 << 16);         // atomic_barrier_addr=0; tensor_dim0[15:0] = 16
    g1[2] = (KCHUNK << 16);     // tensor_dim0[31:16]=0;  tensor_dim1[15:0] = 128
    g1[3] = (16 << 16);         // tensor_dim1[31:16]=0;  tile_dim0 = 16
    g1[4] = KCHUNK;             // tile_dim1 = 128; tile_dim2 = 0
    g1[5] = 16;                 // tensor_dim0_stride[31:0] = 16
    g1[6] = (16 << 16);         // stride0[47:32]=0; tensor_dim1_stride[15:0] = 16
    g1[7] = 0;                  // tensor_dim1_stride[47:16] = 0

    const v4i z4 = {0, 0, 0, 0};              // groups 2/3 unused (2-D tensor)
    const v8i z8 = {0, 0, 0, 0, 0, 0, 0, 0};  // extra group (clang-23 arity)
    __builtin_amdgcn_tensor_load_to_lds(g0, g1, z4, z4, z8, /*cpol=*/0);
}

// ---------------------------------------------------------------------------
// Kernel 1: each block accumulates a partial [64,16] tile over a strided set
// of 128-voxel chunks. 4 waves/block; wave w owns rays [16w,16w+16).
// A-tile (kern weights) computed in-registers as f16; B-tile (mem) TDM-staged
// f32 in double-buffered LDS, converted to f16 at read. One
// v_wmma_f32_16x16x32_f16 per 32-voxel K-slab per wave.
// ---------------------------------------------------------------------------
__global__ __launch_bounds__(128) void tube_gemm_partials(
    const float* __restrict__ origin,   // [64,3]
    const float* __restrict__ dir,      // [64,3]
    const float* __restrict__ mem,      // [N,16] f32
    float* __restrict__ partials,       // [G,64,16]
    int G)
{
    __shared__ float smem[2][KCHUNK * CCH];   // 2 x 8 KB double buffer

    const int tid  = threadIdx.x;
    const int lane = tid & 31;
    const int wav  = tid >> 5;     // 0..3 : which 16-ray tile
    const int h    = lane >> 4;    // wave half (K-split per WMMA layout)
    const int cl   = lane & 15;    // row (A) / column (B,D) within tile
    const int ray  = (wav << 4) | cl;

    // Per-ray scalars (halves of the wave duplicate the same 16 rays).
    const float ox  = origin[ray*3+0], oy = origin[ray*3+1], oz = origin[ray*3+2];
    const float rdx = dir[ray*3+0],    rdy = dir[ray*3+1],   rdz = dir[ray*3+2];
    const float od   = ox*rdx + oy*rdy + oz*rdz;
    const float d2   = rdx*rdx + rdy*rdy + rdz*rdz;
    const float d2m2 = d2 - 2.0f;

    v8f acc = {};  // 16x16 f32 accumulator tile for this wave

    const unsigned lds0 = (unsigned)(uintptr_t)(&smem[0][0]);
    const unsigned lds1 = (unsigned)(uintptr_t)(&smem[1][0]);

    // Prologue: kick off the first tile into buffer 0.
    const int ci0 = blockIdx.x;
    if (wav == 0 && ci0 < NCHUNKS)
        tdm_load_tile(mem + (size_t)ci0 * (KCHUNK * CCH), lds0);

    int p = 0;
    for (int ci = ci0; ci < NCHUNKS; ci += G, p ^= 1) {
        // Wait for the in-flight TDM tile, then make it visible to all waves.
        if (wav == 0) __builtin_amdgcn_s_wait_tensorcnt(0);
        __syncthreads();

        // Overlap: start fetching the next chunk into the other buffer.
        if (wav == 0 && ci + G < NCHUNKS)
            tdm_load_tile(mem + (size_t)(ci + G) * (KCHUNK * CCH),
                          p ? lds0 : lds1);

        // Deep L2 prefetch (two chunks ahead).
        if (ci + 2*G < NCHUNKS)
            __builtin_prefetch(mem + (size_t)(ci + 2*G) * (KCHUNK * CCH), 0, 1);

        // Chunk-constant geometry: n0 = ci*128 -> x = ci>>7, y = ci&127, z=0..127.
        const float xf   = (float)(ci >> 7);
        const float yf   = (float)(ci & 127);
        const float txy  = xf*rdx + yf*rdy - od;     // t = txy + z*rdz
        const float gx   = xf - ox, gy = yf - oy;
        const float dxy2 = gx*gx + gy*gy;            // |dx|^2 = dxy2 + (z-oz)^2

        const float* __restrict__ buf = &smem[p][0];

        #pragma unroll
        for (int c4 = 0; c4 < 4; ++c4) {
            const int zbase = c4 * 32;

            // ---- A tile: 16 kern weights per lane (ISA 16-bit A 16x32 layout)
            v16h a;
            #pragma unroll
            for (int i = 0; i < 16; ++i) {
                const int   koff = (i < 8) ? (h*8 + i) : (8 + h*8 + i);
                const float zf   = (float)(zbase + koff);
                const float t    = fmaf(zf, rdz, txy);
                const float gz   = zf - oz;
                const float dd   = fmaf(gz, gz, dxy2);          // |dx|^2
                const float r2   = fmaf(t*t, d2m2, dd);         // dx2 + t^2(d2-2)
                const float arg  = fmaf(-2.0f, r2, -0.5f * fabsf(t));
                a[i] = (_Float16)__expf(arg);
            }

            // ---- B tile from LDS (lane holds column cl, K-half h;
            //      element i -> K = h*8 + i (i<8), 16 + h*8 + (i-8) (i>=8) within the 32-slab)
            v16h b;
            #pragma unroll
            for (int i = 0; i < 16; ++i) {
                const int k = zbase + ((i < 8) ? (h*8 + i) : (8 + h*8 + i));
                b[i] = (_Float16)buf[k*CCH + cl];
            }

            acc = __builtin_amdgcn_wmma_f32_16x16x32_f16(
                      /*neg_a=*/false, a, /*neg_b=*/false, b,
                      /*c_mod=*/(short)0, acc,
                      /*reuse_a=*/false, /*reuse_b=*/false);
        }
        // All waves done reading buf[p] before it can be refilled next round.
        __syncthreads();
    }

    // Write this block's partial tile. C/D layout: lane group h -> rows v+8h.
    float* pout = partials + (size_t)blockIdx.x * (BRAYS*CCH);
    #pragma unroll
    for (int v = 0; v < 8; ++v) {
        const int row = (wav << 4) + v + (h << 3);
        pout[row*CCH + cl] = acc[v];
    }
}

// ---------------------------------------------------------------------------
// Kernel 2: deterministic serial reduction of G partial tiles -> d_out.
// ---------------------------------------------------------------------------
__global__ void reduce_partials(const float* __restrict__ partials,
                                float* __restrict__ out, int G)
{
    const int i = blockIdx.x * blockDim.x + threadIdx.x;
    if (i >= BRAYS*CCH) return;
    float s = 0.0f;
    for (int g = 0; g < G; ++g)
        s += partials[(size_t)g * (BRAYS*CCH) + i];
    out[i] = s;
}

extern "C" void kernel_launch(void* const* d_in, const int* in_sizes, int n_in,
                              void* d_out, int out_size, void* d_ws, size_t ws_size,
                              hipStream_t stream) {
    const float* origin = (const float*)d_in[0];   // [64,3]
    const float* dir    = (const float*)d_in[1];   // [64,3]
    const float* mem    = (const float*)d_in[2];   // [128,128,128,16]
    float* out  = (float*)d_out;                   // [64,16]
    float* part = (float*)d_ws;

    const size_t per_block = (size_t)BRAYS * CCH * sizeof(float);  // 4 KB
    int G = (int)(ws_size / per_block);
    if (G > 1024) G = 1024;
    if (G < 1)    G = 1;

    tube_gemm_partials<<<G, 128, 0, stream>>>(origin, dir, mem, part, G);
    reduce_partials<<<(BRAYS*CCH + 255)/256, 256, 0, stream>>>(part, out, G);
}
